// DCNv2_10806137717164
// MI455X (gfx1250) — matrix operand order repaired and checked by reading, not compiled, not run on hardware
//
#include <hip/hip_runtime.h>

typedef unsigned int uint;
typedef unsigned short ushort;
typedef __attribute__((ext_vector_type(8)))  float  v8f;
typedef __attribute__((ext_vector_type(8)))  __bf16 v8bf;
typedef __attribute__((ext_vector_type(16))) __bf16 v16bf;

#define BB    4
#define CCH   256
#define HH    80
#define WWD   80
#define HWSZ  6400
#define MTOT  25600      // B*H*W
#define KTOT  2304       // 9 * C
#define NTAPS 9

// ---------------- CDNA5 async global->LDS copies (ASYNCcnt path) -------------
#if defined(__AMDGCN__) && __has_builtin(__builtin_amdgcn_global_load_async_to_lds_b128) && \
    __has_builtin(__builtin_amdgcn_global_load_async_to_lds_b64)
#define HAVE_ASYNC_LDS 1
#else
#define HAVE_ASYNC_LDS 0
#endif

#if HAVE_ASYNC_LDS
typedef int v4i __attribute__((vector_size(16)));
typedef int v2i __attribute__((vector_size(8)));
typedef __attribute__((address_space(1))) v4i gv4i;   // global
typedef __attribute__((address_space(3))) v4i lv4i;   // LDS
typedef __attribute__((address_space(1))) v2i gv2i;
typedef __attribute__((address_space(3))) v2i lv2i;
#endif

__device__ __forceinline__ void async_copy16(const void* g, void* l) {
#if HAVE_ASYNC_LDS
  __builtin_amdgcn_global_load_async_to_lds_b128((gv4i*)(size_t)g, (lv4i*)l, 0, 0);
#else
  *(uint4*)l = *(const uint4*)g;
#endif
}
__device__ __forceinline__ void async_copy8(const void* g, void* l) {
#if HAVE_ASYNC_LDS
  __builtin_amdgcn_global_load_async_to_lds_b64((gv2i*)(size_t)g, (lv2i*)l, 0, 0);
#else
  *(uint2*)l = *(const uint2*)g;
#endif
}
template <int N>
__device__ __forceinline__ void async_wait() {
#if HAVE_ASYNC_LDS
#if __has_builtin(__builtin_amdgcn_s_wait_asynccnt)
  __builtin_amdgcn_s_wait_asynccnt(N);
#else
  asm volatile("s_wait_asynccnt %0" ::"i"(N) : "memory");
#endif
#endif
}

// ---------------- scalar helpers ----------------
__device__ __forceinline__ ushort f2bf(float f) {
  uint u = __float_as_uint(f);
  uint r = u + 0x7FFFu + ((u >> 16) & 1u);   // round-to-nearest-even
  return (ushort)(r >> 16);
}
__device__ __forceinline__ float bf_lo(uint p) { return __uint_as_float(p << 16); }
__device__ __forceinline__ float bf_hi(uint p) { return __uint_as_float(p & 0xFFFF0000u); }
__device__ __forceinline__ uint pack_bf2(float a, float b) {
  return (uint)f2bf(a) | ((uint)f2bf(b) << 16);
}

// A fragment (16x32 bf16): lane L -> row = L%16, K-half = L/16.
// Lane holds K=[16*kh..16*kh+15] contiguous (matches B fragment)
__device__ __forceinline__ v16bf lds_afrag(const ushort* base, int lane) {
  int r = lane & 15, kh = lane >> 4;
  v8bf lo = *(const v8bf*)(base + r * 32 + kh * 16);
  v8bf hi = *(const v8bf*)(base + r * 32 + kh * 16 + 8);
  v16bf a;
#pragma unroll
  for (int i = 0; i < 8; ++i) { a[i] = lo[i]; a[i + 8] = hi[i]; }
  return a;
}
// B fragment (32x16 bf16) from LDS column-major [n][k], row stride 32:
// lane L -> col = L%16, K=[16*(L/16)..+15] contiguous -> one 32B read.
__device__ __forceinline__ v16bf lds_bfrag(const ushort* base, int lane) {
  int n = lane & 15, kh = lane >> 4;
  return *(const v16bf*)(base + n * 32 + kh * 16);
}
__device__ __forceinline__ v8f wmma_bf16(v16bf a, v16bf b, v8f c) {
  return __builtin_amdgcn_wmma_f32_16x16x32_bf16(false, a, false, b, (short)0, c,
                                                 false, false);
}

// ---------------- prep kernels ----------------
__global__ __launch_bounds__(256) void k_pack_x(const float* __restrict__ x,
                                                ushort* __restrict__ xh) {
  int i = blockIdx.x * 256 + threadIdx.x;
  if (i >= BB * CCH * HWSZ) return;
  int b  = i / (CCH * HWSZ);
  int r  = i % (CCH * HWSZ);
  int c  = r / HWSZ;
  int hw = r % HWSZ;
  xh[(size_t)(b * HWSZ + hw) * CCH + c] = f2bf(x[i]);
}
__global__ __launch_bounds__(256) void k_pack_w(const float* __restrict__ w,
                                                ushort* __restrict__ wt) {
  int i = blockIdx.x * 256 + threadIdx.x;
  if (i >= CCH * KTOT) return;
  int n = i / KTOT, kk = i % KTOT;
  int t = kk / CCH, c = kk % CCH;
  wt[i] = f2bf(w[(size_t)(n * CCH + c) * 9 + t]);
}
__global__ __launch_bounds__(256) void k_pack_wom(const float* __restrict__ w,
                                                  ushort* __restrict__ womt) {
  int i = blockIdx.x * 256 + threadIdx.x;
  if (i >= 32 * KTOT) return;
  int n = i / KTOT, kk = i % KTOT;
  int t = kk / CCH, c = kk % CCH;
  float v = (n < 27) ? w[(size_t)(n * CCH + c) * 9 + t] : 0.f;
  womt[i] = f2bf(v);
}
__global__ __launch_bounds__(256) void k_zero(float* __restrict__ p) {
  int i = blockIdx.x * 256 + threadIdx.x;
  if (i < 512) p[i] = 0.f;
}

// -------- offset/mask conv (implicit GEMM, M-tile=128, N=32) --------
// Epilogue emits per-(pixel,tap) gather tables:
// samp[(m*9+t)*8] = {idx0..3 (u32 pixel idx into x_hwc), w0..3 (f32, incl. mask)}
__global__ __launch_bounds__(256) void k_offconv(const ushort* __restrict__ xh,
                                                 const ushort* __restrict__ womt,
                                                 const float* __restrict__ b_om,
                                                 uint* __restrict__ samp) {
  __shared__ __align__(32) ushort At[128][32];
  __shared__ __align__(32) ushort Bt[32][32];
  __shared__ float Et[128][32];

  int tid = threadIdx.x;
  int wv = tid >> 5, lane = tid & 31;
  int m0 = blockIdx.x * 128;

  int p = tid >> 1, half = tid & 1;
  int m = m0 + p;
  int bimg = m / HWSZ, hw = m % HWSZ;
  int h = hw / WWD, w = hw % WWD;

  int nB = tid >> 3, kqB = (tid & 7) << 2;   // B-tile copy role
  const ushort* wom_row = womt + (size_t)nB * KTOT + kqB;

  v8f acc0 = {}, acc1 = {};

  for (int kb = 0; kb < KTOT; kb += 32) {
    int t = kb >> 8;                          // tap fixed within chunk (C=256)
    int cbase = kb & 255;
    int ky = t / 3, kx = t % 3;
    int yy = h - 1 + ky, xx = w - 1 + kx;
    uint4* dstA = (uint4*)(&At[p][half * 16]);
    if (yy >= 0 && yy < HH && xx >= 0 && xx < WWD) {
      const ushort* src = xh +
          ((size_t)(bimg * HWSZ + yy * WWD + xx)) * CCH + cbase + half * 16;
      async_copy16(src, dstA);
      async_copy16(src + 8, dstA + 1);
    } else {
      uint4 z = {0, 0, 0, 0};
      dstA[0] = z; dstA[1] = z;
    }
    async_copy8(wom_row + kb, &Bt[nB][kqB]);
    async_wait<0>();
    __syncthreads();
    v16bf a  = lds_afrag(&At[wv * 16][0], lane);
    v16bf b0 = lds_bfrag(&Bt[0][0], lane);
    v16bf b1 = lds_bfrag(&Bt[16][0], lane);
    acc0 = wmma_bf16(a, b0, acc0);
    acc1 = wmma_bf16(a, b1, acc1);
    __syncthreads();
  }

  // accum tile -> Et[pixel][channel]
  int n0 = lane & 15;
  int mr = wv * 16 + ((lane >> 4) << 3);
  float bo0 = b_om[n0];
  float bo1 = (16 + n0 < 27) ? b_om[16 + n0] : 0.f;
#pragma unroll
  for (int k = 0; k < 8; ++k) {
    Et[mr + k][n0]      = acc0[k] + bo0;
    Et[mr + k][16 + n0] = acc1[k] + bo1;
  }
  __syncthreads();

  if (tid < 128) {
    const float* E = &Et[tid][0];
    int mg = m0 + tid;
    int bi = mg / HWSZ;
    int hwg = mg % HWSZ;
    int hg = hwg / WWD, wg = hwg % WWD;
#pragma unroll
    for (int t = 0; t < NTAPS; ++t) {
      float dy = E[2 * t], dx = E[2 * t + 1];
      float mk = 1.f / (1.f + __expf(-E[18 + t]));
      int ky = t / 3, kx = t % 3;
      float ys = dy + (float)(hg - 1 + ky);
      float xs = dx + (float)(wg - 1 + kx);
      float y0f = floorf(ys), x0f = floorf(xs);
      int y0 = (int)y0f, x0 = (int)x0f;
      float fy = ys - y0f, fx = xs - x0f;
      int y1 = y0 + 1, x1 = x0 + 1;
      bool vy0 = (y0 >= 0) && (y0 < HH), vy1 = (y1 >= 0) && (y1 < HH);
      bool vx0 = (x0 >= 0) && (x0 < WWD), vx1 = (x1 >= 0) && (x1 < WWD);
      int cy0 = min(max(y0, 0), HH - 1), cy1 = min(max(y1, 0), HH - 1);
      int cx0 = min(max(x0, 0), WWD - 1), cx1 = min(max(x1, 0), WWD - 1);
      uint bbs = (uint)(bi * HWSZ);
      uint4 iv = { bbs + (uint)(cy0 * WWD + cx0), bbs + (uint)(cy0 * WWD + cx1),
                   bbs + (uint)(cy1 * WWD + cx0), bbs + (uint)(cy1 * WWD + cx1) };
      float w00 = (vy0 && vx0) ? (1.f - fy) * (1.f - fx) * mk : 0.f;
      float w01 = (vy0 && vx1) ? (1.f - fy) * fx * mk : 0.f;
      float w10 = (vy1 && vx0) ? fy * (1.f - fx) * mk : 0.f;
      float w11 = (vy1 && vx1) ? fy * fx * mk : 0.f;
      uint4 wvv = { __float_as_uint(w00), __float_as_uint(w01),
                    __float_as_uint(w10), __float_as_uint(w11) };
      uint4* dst = (uint4*)(samp + ((size_t)mg * NTAPS + t) * 8);
      dst[0] = iv; dst[1] = wvv;
    }
  }
}

// -------- main DCN implicit GEMM: M-tile=64, N=256, K chunks of 32 --------
// Double-buffered B tile fed by async global->LDS copies overlapped with the
// bilinear-gather A-tile build and the WMMAs.
__global__ __launch_bounds__(256) void k_dcngemm(const ushort* __restrict__ xh,
                                                 const ushort* __restrict__ wt,
                                                 const uint* __restrict__ samp,
                                                 const float* __restrict__ bias,
                                                 float* __restrict__ y,
                                                 float* __restrict__ sums,
                                                 float* __restrict__ sumsq) {
  __shared__ __align__(32) ushort At[64][32];
  __shared__ __align__(32) ushort Bt[2][256][32];
  __shared__ uint Smp[64 * NTAPS * 8];

  int tid = threadIdx.x;
  int wv = tid >> 5, lane = tid & 31;
  int m0 = blockIdx.x * 64;

  // async preload: gather tables (18KB) + B chunk 0 (thread tid owns row tid)
  {
    const uint4* sg = (const uint4*)(samp + (size_t)m0 * NTAPS * 8);
    uint4* sl = (uint4*)Smp;
    for (int i = tid; i < 64 * NTAPS * 2; i += 256)
      async_copy16(&sg[i], &sl[i]);
  }
  const ushort* wrow = wt + (size_t)tid * KTOT;
#pragma unroll
  for (int q = 0; q < 4; ++q)
    async_copy16(wrow + q * 8, &Bt[0][tid][q * 8]);
  async_wait<4>();   // Smp complete; B0 may still be in flight
  __syncthreads();

  v8f acc[2][4] = {};
  int p = tid >> 2;               // 0..63 : pixel for A build
  int cg = (tid & 3) << 3;        // channel offset within 32-chunk
  int buf = 0;

  for (int kb = 0; kb < KTOT; kb += 32) {
    bool more = (kb + 32) < KTOT;
    if (more) {                   // kick off next B chunk into other buffer
#pragma unroll
      for (int q = 0; q < 4; ++q)
        async_copy16(wrow + kb + 32 + q * 8, &Bt[buf ^ 1][tid][q * 8]);
    }
    // ---- build A tile: a[m][k] = sum_i w_i * x_hwc[idx_i][c] (bilinear+mask)
    int t = kb >> 8;
    int cbase = kb & 255;
    const uint* sp = &Smp[(p * NTAPS + t) * 8];
    float a0 = 0.f, a1 = 0.f, a2 = 0.f, a3 = 0.f,
          a4 = 0.f, a5 = 0.f, a6 = 0.f, a7 = 0.f;
#pragma unroll
    for (int i = 0; i < 4; ++i) {
      uint idx = sp[i];
      float cw = __uint_as_float(sp[4 + i]);
      uint4 r = *(const uint4*)(xh + (size_t)idx * CCH + cbase + cg);
      a0 += cw * bf_lo(r.x); a1 += cw * bf_hi(r.x);
      a2 += cw * bf_lo(r.y); a3 += cw * bf_hi(r.y);
      a4 += cw * bf_lo(r.z); a5 += cw * bf_hi(r.z);
      a6 += cw * bf_lo(r.w); a7 += cw * bf_hi(r.w);
    }
    uint4 packed = { pack_bf2(a0, a1), pack_bf2(a2, a3),
                     pack_bf2(a4, a5), pack_bf2(a6, a7) };
    *(uint4*)(&At[p][cg]) = packed;

    if (more) async_wait<4>();    // current B done; next-B copies in flight
    else      async_wait<0>();
    __syncthreads();

    // ---- WMMA: wave owns 2 M-subtiles x 4 N-subtiles, A-fragments reused
    v16bf af0 = lds_afrag(&At[((wv & 1) * 2 + 0) * 16][0], lane);
    v16bf af1 = lds_afrag(&At[((wv & 1) * 2 + 1) * 16][0], lane);
#pragma unroll
    for (int j = 0; j < 4; ++j) {
      int nt = (wv >> 1) * 4 + j;
      v16bf bf = lds_bfrag(&Bt[buf][nt * 16][0], lane);
      acc[0][j] = wmma_bf16(af0, bf, acc[0][j]);
      acc[1][j] = wmma_bf16(af1, bf, acc[1][j]);
    }
    __syncthreads();
    buf ^= 1;
  }

  // ---- epilogue: +bias, two b128 stores of 8 consecutive pixels per lane,
  //      fused BN statistics via per-channel f32 atomics.
  int nloc = lane & 15;
  int mrow = (lane >> 4) << 3;
#pragma unroll
  for (int mi = 0; mi < 2; ++mi) {
    int mloc = ((wv & 1) * 2 + mi) * 16 + mrow;
    int mg = m0 + mloc;
    int bi = mg / HWSZ, hwg = mg % HWSZ;
#pragma unroll
    for (int j = 0; j < 4; ++j) {
      int n = ((wv >> 1) * 4 + j) * 16 + nloc;
      float bv = bias[n];
      float s = 0.f, ss = 0.f, v[8];
#pragma unroll
      for (int k = 0; k < 8; ++k) {
        float e = acc[mi][j][k] + bv;
        v[k] = e; s += e; ss += e * e;
      }
      float4* dst = (float4*)(y + ((size_t)(bi * CCH + n)) * HWSZ + hwg);
      dst[0] = make_float4(v[0], v[1], v[2], v[3]);
      dst[1] = make_float4(v[4], v[5], v[6], v[7]);
      __hip_atomic_fetch_add(&sums[n],  s,  __ATOMIC_RELAXED, __HIP_MEMORY_SCOPE_AGENT);
      __hip_atomic_fetch_add(&sumsq[n], ss, __ATOMIC_RELAXED, __HIP_MEMORY_SCOPE_AGENT);
    }
  }
}

// ---------------- BN finalize ----------------
__global__ __launch_bounds__(256) void k_stats(const float* __restrict__ sums,
                                               const float* __restrict__ sumsq,
                                               const float* __restrict__ gamma,
                                               const float* __restrict__ beta,
                                               float* __restrict__ scale,
                                               float* __restrict__ shift) {
  int c = threadIdx.x;
  if (c >= CCH) return;
  const float inv = 1.f / (float)MTOT;
  float mean = sums[c] * inv;
  float var  = sumsq[c] * inv - mean * mean;
  float sc   = gamma[c] * rsqrtf(var + 1e-5f);
  scale[c] = sc;
  shift[c] = beta[c] - mean * sc;
}
__global__ __launch_bounds__(256) void k_bnact(float* __restrict__ y,
                                               const float* __restrict__ scale,
                                               const float* __restrict__ shift) {
  size_t i = (size_t)blockIdx.x * 256 + threadIdx.x;
  if (i >= (size_t)BB * CCH * HWSZ) return;
  int c = (int)((i / HWSZ) % CCH);
  float v = y[i] * scale[c] + shift[c];
  y[i] = v / (1.f + __expf(-v));   // SiLU
}

// ---------------- host ----------------
extern "C" void kernel_launch(void* const* d_in, const int* in_sizes, int n_in,
                              void* d_out, int out_size, void* d_ws, size_t ws_size,
                              hipStream_t stream) {
  const float* x      = (const float*)d_in[0];
  const float* w_om   = (const float*)d_in[1];
  const float* b_om   = (const float*)d_in[2];
  const float* weight = (const float*)d_in[3];
  const float* bias   = (const float*)d_in[4];
  const float* gamma  = (const float*)d_in[5];
  const float* beta   = (const float*)d_in[6];
  float* out = (float*)d_out;

  char* ws = (char*)d_ws;
  size_t off = 0;
  auto alloc = [&](size_t bytes) -> void* {
    void* p = ws + off;
    off += (bytes + 255) & ~(size_t)255;
    return p;
  };
  ushort* xh   = (ushort*)alloc((size_t)MTOT * CCH * 2);        // 13.1 MB
  ushort* wt   = (ushort*)alloc((size_t)CCH * KTOT * 2);        // 1.18 MB
  ushort* womt = (ushort*)alloc((size_t)32 * KTOT * 2);         // 147 KB
  uint*   samp = (uint*)alloc((size_t)MTOT * NTAPS * 8 * 4);    // 7.37 MB
  float*  sums = (float*)alloc(512 * 4);                        // sums + sumsq
  float*  scl  = (float*)alloc(256 * 4);
  float*  shf  = (float*)alloc(256 * 4);

  k_pack_x  <<<(MTOT * CCH + 255) / 256, 256, 0, stream>>>(x, xh);
  k_pack_w  <<<(CCH * KTOT + 255) / 256, 256, 0, stream>>>(weight, wt);
  k_pack_wom<<<(32 * KTOT + 255) / 256, 256, 0, stream>>>(w_om, womt);
  k_zero    <<<2, 256, 0, stream>>>(sums);
  k_offconv <<<MTOT / 128, 256, 0, stream>>>(xh, womt, b_om, samp);
  k_dcngemm <<<MTOT / 64, 256, 0, stream>>>(xh, wt, samp, bias, out,
                                            sums, sums + 256);
  k_stats   <<<1, 256, 0, stream>>>(sums, sums + 256, gamma, beta, scl, shf);
  k_bnact   <<<(MTOT * CCH) / 256, 256, 0, stream>>>(out, scl, shf);
}
